// MultiHeadSelfAttention_37967510896884
// MI455X (gfx1250) — matrix operand order repaired and checked by reading, not compiled, not run on hardware
//
#include <hip/hip_runtime.h>
#include <hip/hip_bf16.h>
#include <math.h>

// Problem dims (fixed by reference)
#define B_  4
#define S_  2048
#define D_  1024
#define H_  16
#define E_  64

typedef __attribute__((ext_vector_type(16))) __bf16 v16bf;
typedef __attribute__((ext_vector_type(8)))  __bf16 v8bf;
typedef __attribute__((ext_vector_type(8)))  float  v8f;

static __device__ __forceinline__ __bf16 f2bf(float x) { return (__bf16)x; }

union BF16x16 { v16bf v; v8bf h[2]; };

// ---------------------------------------------------------------------------
// WMMA fragment loaders (wave32, V_WMMA_F32_16X16X32_BF16)
// A (16x32, MxK): lane row M = lane&15 ; ko = (lane>>4)*8
//   element j -> K = (j/8)*16 + ko + (j%8)   (per ISA 16-bit A table)
// B (32x16, KxN): lane col N = lane&15 ; ko = (lane>>4)*8, element j -> K = (j/8)*16 + ko + (j%8)
//   sourced from a layout where BT[n][k] is contiguous in k (we pre-transpose)
// ---------------------------------------------------------------------------
static __device__ __forceinline__ v16bf load_a_bf16(const __bf16* base, int stride, int lane) {
  const __bf16* p = base + (size_t)(lane & 15) * stride + ((lane >> 4) * 8);
  BF16x16 u;
  u.h[0] = *(const v8bf*)(p);        // K group 0..7 (+ko)
  u.h[1] = *(const v8bf*)(p + 16);   // K group 16..23 (+ko)
  return u.v;
}

static __device__ __forceinline__ v16bf load_a_f32(const float* base, int stride, int lane) {
  const float* p = base + (size_t)(lane & 15) * stride + ((lane >> 4) * 8);
  v16bf a;
  #pragma unroll
  for (int j = 0; j < 8; ++j) {
    a[j]     = f2bf(p[j]);
    a[8 + j] = f2bf(p[16 + j]);
  }
  return a;
}

static __device__ __forceinline__ v16bf load_b_bf16(const __bf16* baseT, int stride, int lane) {
  const __bf16* p = baseT + (size_t)(lane & 15) * stride + ((lane >> 4) * 8);
  BF16x16 u;
  u.h[0] = *(const v8bf*)(p);
  u.h[1] = *(const v8bf*)(p + 16);
  return u.v;
}

static __device__ __forceinline__ v8f wmma_bf16(v16bf a, v16bf b, v8f c) {
  return __builtin_amdgcn_wmma_f32_16x16x32_bf16(false, a, false, b, (short)0, c, false, false);
}

// ---------------------------------------------------------------------------
// Tensor Data Mover: build D# per CDNA5 ISA 8.3/8.4 and issue TENSOR_LOAD_TO_LDS
// This toolchain exposes the 6-arg builtin:
//   (u32x4 g0, i32x8 g1, i32x4 g2, i32x4 g3, i32x8 extra, i32 cpol)
// ---------------------------------------------------------------------------
#if defined(__has_builtin)
#if __has_builtin(__builtin_amdgcn_tensor_load_to_lds)
#define HAVE_TDM 1
#endif
#endif

#ifdef HAVE_TDM
typedef __attribute__((ext_vector_type(4))) unsigned int u32x4_t;
typedef __attribute__((ext_vector_type(8))) int i32x8_t;
typedef __attribute__((ext_vector_type(4))) int i32x4_t;

// 2-D tile load (data_size = 8 bytes). 1-D: pass tile_d1 = 0, tensor_d1 = 1.
// Dims/strides in 8-byte units.
static __device__ __forceinline__ void tdm_load(unsigned int lds_off, unsigned long long ga,
                                                int tensor_d0, int tensor_d1,
                                                int tile_d0, int tile_d1, int d0_stride) {
  u32x4_t g0;
  g0[0] = 1u;                                               // count=1, no gather
  g0[1] = lds_off;                                          // lds_addr (bytes)
  g0[2] = (unsigned int)ga;                                 // global_addr[31:0]
  g0[3] = ((unsigned int)(ga >> 32) & 0x01FFFFFFu) | (2u << 30); // addr[56:32] | type=2
  i32x8_t g1;
  g1[0] = 3 << 16;                                          // data_size = 8B; mask=0
  g1[1] = (tensor_d0 & 0xFFFF) << 16;                       // tensor_dim0[15:0]
  g1[2] = ((tensor_d0 >> 16) & 0xFFFF) | ((tensor_d1 & 0xFFFF) << 16);
  g1[3] = ((tensor_d1 >> 16) & 0xFFFF) | ((tile_d0 & 0xFFFF) << 16);
  g1[4] = (tile_d1 & 0xFFFF);                               // tile_dim1 ; tile_dim2=0
  g1[5] = d0_stride;                                        // tensor_dim0_stride[31:0]
  g1[6] = 0;
  g1[7] = 0;
  i32x4_t z4 = {};
  i32x8_t z8 = {};
  __builtin_amdgcn_tensor_load_to_lds(g0, g1, z4, z4, z8, 0);
}
#endif

// ---------------------------------------------------------------------------
// K0: transpose + f32->bf16 convert.  out[b][c][r] = in[b][r][c]
// ---------------------------------------------------------------------------
__global__ void k_transpose_bf16(const float* __restrict__ in, __bf16* __restrict__ out,
                                 int batch, int R, int C) {
  size_t i = (size_t)blockIdx.x * blockDim.x + threadIdx.x;
  size_t total = (size_t)batch * R * C;
  if (i >= total) return;
  int c = (int)(i % C);
  size_t t = i / C;
  int r = (int)(t % R);
  int b = (int)(t / R);
  out[((size_t)b * C + c) * R + r] = f2bf(in[i]);
}

// ---------------------------------------------------------------------------
// K1: per-head projection GEMM.  X:[B,S,D] f32, WT:[H,E,D] bf16, bias:[H,E] f32
// out: transposeOut==0 -> [B,H,S,E] bf16 ; ==1 -> [B,H,E,S] bf16 (V^T)
// ---------------------------------------------------------------------------
__global__ void __launch_bounds__(256)
k_qkv_proj(const float* __restrict__ X, const __bf16* __restrict__ WT,
           const float* __restrict__ bias, __bf16* __restrict__ out, int transposeOut) {
  const int lane = threadIdx.x & 31;
  const int wave = threadIdx.x >> 5;
  const int sb   = blockIdx.x * 128 + wave * 16;
  const int b    = blockIdx.y / H_;
  const int h    = blockIdx.y % H_;

  const float*  xrow = X + ((size_t)b * S_ + sb) * D_;
  const __bf16* wrow = WT + (size_t)h * E_ * D_;

  v8f acc[4];
  #pragma unroll
  for (int nt = 0; nt < 4; ++nt) { v8f z = {}; acc[nt] = z; }

  for (int kb = 0; kb < D_; kb += 32) {
    __builtin_prefetch(xrow + kb + 64, 0, 1);
    v16bf a = load_a_f32(xrow + kb, D_, lane);
    #pragma unroll
    for (int nt = 0; nt < 4; ++nt) {
      v16bf bm = load_b_bf16(wrow + (size_t)(nt * 16) * D_ + kb, D_, lane);
      acc[nt] = wmma_bf16(a, bm, acc[nt]);
    }
  }

  const int halfsel = lane >> 4;
  const int ncol0   = lane & 15;
  #pragma unroll
  for (int nt = 0; nt < 4; ++nt) {
    int e = nt * 16 + ncol0;
    float bv = bias[h * E_ + e];
    #pragma unroll
    for (int r = 0; r < 8; ++r) {
      int srow = sb + halfsel * 8 + r;
      float val = acc[nt][r] + bv;
      size_t o = transposeOut
          ? (((size_t)(b * H_ + h) * E_ + e) * S_ + srow)
          : (((size_t)(b * H_ + h) * S_ + srow) * E_ + e);
      out[o] = f2bf(val);
    }
  }
}

// ---------------------------------------------------------------------------
// K2: causal flash attention.  Q,K:[B,H,S,E] bf16, VT:[B,H,E,S] bf16
// out ATT:[B,S,D] bf16.  Per 64-key chunk, wave 0 DMAs the K tile (8KB,
// contiguous) and V^T tile (64 rows x 128B, stride S*2B) into LDS via the
// Tensor Data Mover; all 8 waves consume fragments from LDS (removes the 8x
// redundant global fetch).  Online softmax in f32.
// ---------------------------------------------------------------------------
__global__ void __launch_bounds__(256)
k_flash_attn(const __bf16* __restrict__ Q, const __bf16* __restrict__ K,
             const __bf16* __restrict__ VT, __bf16* __restrict__ ATT) {
  __shared__ __align__(16) __bf16 ldsK[64 * 64];     // [key][e]   8 KB
  __shared__ __align__(16) __bf16 ldsV[64 * 64];     // [e][key]   8 KB
  __shared__ __align__(16) __bf16 lds_p[8][16][72];  // per-wave P tile, padded

  const int lane = threadIdx.x & 31;
  const int wave = threadIdx.x >> 5;
  const int qb   = blockIdx.x * 128;
  const int qt   = qb + wave * 16;
  const int bh   = blockIdx.y;
  const int b    = bh / H_;
  const int h    = bh % H_;

  const __bf16* qbase = Q  + ((size_t)bh * S_ + qt) * E_;
  const __bf16* kbase = K  + (size_t)bh * S_ * E_;
  const __bf16* vbase = VT + (size_t)bh * E_ * S_;

  v16bf aq0 = load_a_bf16(qbase + 0,  E_, lane);
  v16bf aq1 = load_a_bf16(qbase + 32, E_, lane);

  v8f ov[4];
  #pragma unroll
  for (int nt = 0; nt < 4; ++nt) { v8f z = {}; ov[nt] = z; }
  float mrow[8], lrow[8];
  #pragma unroll
  for (int r = 0; r < 8; ++r) { mrow[r] = -1e30f; lrow[r] = 0.0f; }

  const float scale  = 0.125f;  // 1/sqrt(E)
  const int halfsel  = lane >> 4;
  const int ncol0    = lane & 15;
  const int kend     = qb + 128;  // uniform causal bound for the block

  for (int kc = 0; kc < kend; kc += 64) {
    // ---- stage K chunk [kc..kc+63][0..63] and V^T chunk into LDS
#ifdef HAVE_TDM
    if (wave == 0) {
      unsigned long long gaK = (unsigned long long)(uintptr_t)(kbase + (size_t)kc * E_);
      // 1-D: 64*64 bf16 = 1024 x 8B contiguous
      tdm_load((unsigned int)(uintptr_t)ldsK, gaK, 1024, 1, 1024, 0, 1024);
      unsigned long long gaV = (unsigned long long)(uintptr_t)(vbase + kc);
      // 2-D: 64 rows (e) of 16 x 8B, row stride S*2B = 512 x 8B
      tdm_load((unsigned int)(uintptr_t)ldsV, gaV, 16, 64, 16, 64, (S_ * 2) / 8);
      __builtin_amdgcn_s_wait_tensorcnt(0);
    }
    __syncthreads();
#else
    for (int i = (int)threadIdx.x; i < 64 * 64 / 2; i += 256) {
      ((unsigned int*)ldsK)[i] =
          ((const unsigned int*)(kbase + (size_t)kc * E_))[i];
      int e = i >> 5, c = i & 31;
      ((unsigned int*)ldsV)[i] =
          ((const unsigned int*)(vbase + (size_t)e * S_ + kc))[c];
    }
    __syncthreads();
#endif

    // ---- scores: 16x64 = 4 N-tiles x 2 K-steps of WMMA (K from LDS)
    v8f sc[4];
    #pragma unroll
    for (int nt = 0; nt < 4; ++nt) {
      v8f z = {}; sc[nt] = z;
      const __bf16* kt = ldsK + (nt * 16) * 64;
      sc[nt] = wmma_bf16(aq0, load_b_bf16(kt + 0,  64, lane), sc[nt]);
      sc[nt] = wmma_bf16(aq1, load_b_bf16(kt + 32, 64, lane), sc[nt]);
    }

    // ---- causal mask + per-row running max (rows live in 16-lane halves)
    float newm[8];
    #pragma unroll
    for (int r = 0; r < 8; ++r) {
      int qrow = qt + halfsel * 8 + r;
      float mx = -1e30f;
      #pragma unroll
      for (int nt = 0; nt < 4; ++nt) {
        int kcol = kc + nt * 16 + ncol0;
        float s = sc[nt][r] * scale;
        if (kcol > qrow) s = -1e30f;
        sc[nt][r] = s;
        mx = fmaxf(mx, s);
      }
      #pragma unroll
      for (int off = 1; off < 16; off <<= 1)
        mx = fmaxf(mx, __shfl_xor(mx, off, 32));
      newm[r] = fmaxf(mrow[r], mx);
    }

    // ---- p = exp(s-m), row sums, accumulator rescale, stage P in LDS
    #pragma unroll
    for (int r = 0; r < 8; ++r) {
      float rs = 0.0f;
      #pragma unroll
      for (int nt = 0; nt < 4; ++nt) {
        float p = __expf(sc[nt][r] - newm[r]);
        rs += p;
        lds_p[wave][halfsel * 8 + r][nt * 16 + ncol0] = f2bf(p);
      }
      #pragma unroll
      for (int off = 1; off < 16; off <<= 1)
        rs += __shfl_xor(rs, off, 32);
      float alpha = __expf(mrow[r] - newm[r]);
      lrow[r] = lrow[r] * alpha + rs;
      mrow[r] = newm[r];
      #pragma unroll
      for (int nt = 0; nt < 4; ++nt) ov[nt][r] *= alpha;
    }
    __syncthreads();

    // ---- P (C-layout) -> A-fragments via LDS, then O += P x V (V from LDS)
    v16bf ap0 = load_a_bf16(&lds_p[wave][0][0] + 0,  72, lane);
    v16bf ap1 = load_a_bf16(&lds_p[wave][0][0] + 32, 72, lane);
    #pragma unroll
    for (int nt = 0; nt < 4; ++nt) {
      const __bf16* vt = ldsV + (nt * 16) * 64;
      ov[nt] = wmma_bf16(ap0, load_b_bf16(vt + 0,  64, lane), ov[nt]);
      ov[nt] = wmma_bf16(ap1, load_b_bf16(vt + 32, 64, lane), ov[nt]);
    }
    __syncthreads();  // protect LDS K/V/P before next chunk's DMA overwrites
  }

  // ---- epilogue: O /= l, store bf16 into concat-head layout [B,S,D]
  #pragma unroll
  for (int nt = 0; nt < 4; ++nt) {
    int e = nt * 16 + ncol0;
    #pragma unroll
    for (int r = 0; r < 8; ++r) {
      int srow = qt + halfsel * 8 + r;
      float val = ov[nt][r] / lrow[r];
      ATT[((size_t)b * S_ + srow) * D_ + h * E_ + e] = f2bf(val);
    }
  }
}

// ---------------------------------------------------------------------------
// K3: output projection.  A:[B*S,D] bf16, WT:[D,D]=Wp^T bf16, bias:[D] f32
// ---------------------------------------------------------------------------
__global__ void __launch_bounds__(256)
k_out_proj(const __bf16* __restrict__ A, const __bf16* __restrict__ WT,
           const float* __restrict__ bias, float* __restrict__ out) {
  const int lane = threadIdx.x & 31;
  const int wave = threadIdx.x >> 5;
  const int mb   = blockIdx.x * 128 + wave * 16;
  const int nb   = blockIdx.y * 64;

  const __bf16* arow = A + (size_t)mb * D_;
  v8f acc[4];
  #pragma unroll
  for (int nt = 0; nt < 4; ++nt) { v8f z = {}; acc[nt] = z; }

  for (int kb = 0; kb < D_; kb += 32) {
    __builtin_prefetch(arow + kb + 64, 0, 1);
    v16bf a = load_a_bf16(arow + kb, D_, lane);
    #pragma unroll
    for (int nt = 0; nt < 4; ++nt) {
      v16bf bm = load_b_bf16(WT + (size_t)(nb + nt * 16) * D_ + kb, D_, lane);
      acc[nt] = wmma_bf16(a, bm, acc[nt]);
    }
  }

  const int halfsel = lane >> 4;
  const int ncol0   = lane & 15;
  #pragma unroll
  for (int nt = 0; nt < 4; ++nt) {
    int n = nb + nt * 16 + ncol0;
    float bv = bias[n];
    #pragma unroll
    for (int r = 0; r < 8; ++r) {
      out[(size_t)(mb + halfsel * 8 + r) * D_ + n] = acc[nt][r] + bv;
    }
  }
}

// ---------------------------------------------------------------------------
// Host launcher
// ---------------------------------------------------------------------------
extern "C" void kernel_launch(void* const* d_in, const int* in_sizes, int n_in,
                              void* d_out, int out_size, void* d_ws, size_t ws_size,
                              hipStream_t stream) {
  (void)in_sizes; (void)n_in; (void)out_size; (void)ws_size;
  const float* values  = (const float*)d_in[0];
  const float* keys    = (const float*)d_in[1];
  const float* queries = (const float*)d_in[2];
  const float* Wq = (const float*)d_in[3];
  const float* bq = (const float*)d_in[4];
  const float* Wk = (const float*)d_in[5];
  const float* bk = (const float*)d_in[6];
  const float* Wv = (const float*)d_in[7];
  const float* bv = (const float*)d_in[8];
  const float* Wp = (const float*)d_in[9];
  const float* bp = (const float*)d_in[10];
  float* out = (float*)d_out;

  char* ws = (char*)d_ws;
  const size_t szW  = (size_t)H_ * E_ * D_ * 2;       // 2 MB per QKV weight
  const size_t szWp = (size_t)D_ * D_ * 2;            // 2 MB
  const size_t szQ  = (size_t)B_ * H_ * S_ * E_ * 2;  // 16 MB each
  __bf16* WqT = (__bf16*)(ws);
  __bf16* WkT = (__bf16*)(ws + szW);
  __bf16* WvT = (__bf16*)(ws + 2 * szW);
  __bf16* WpT = (__bf16*)(ws + 3 * szW);
  __bf16* Qb  = (__bf16*)(ws + 3 * szW + szWp);
  __bf16* Kb  = (__bf16*)(ws + 3 * szW + szWp + szQ);
  __bf16* VTb = (__bf16*)(ws + 3 * szW + szWp + 2 * szQ);
  __bf16* ATT = (__bf16*)(ws + 3 * szW + szWp + 3 * szQ);
  // total workspace: 8 MB + 48 MB + 16 MB = 72 MB

  {
    int n = H_ * D_ * E_;
    int blocks = (n + 255) / 256;
    k_transpose_bf16<<<blocks, 256, 0, stream>>>(Wq, WqT, H_, D_, E_);
    k_transpose_bf16<<<blocks, 256, 0, stream>>>(Wk, WkT, H_, D_, E_);
    k_transpose_bf16<<<blocks, 256, 0, stream>>>(Wv, WvT, H_, D_, E_);
    int n2 = D_ * D_;
    k_transpose_bf16<<<(n2 + 255) / 256, 256, 0, stream>>>(Wp, WpT, 1, D_, D_);
  }

  dim3 gqkv(S_ / 128, B_ * H_);
  k_qkv_proj<<<gqkv, 256, 0, stream>>>(queries, WqT, bq, Qb, 0);
  k_qkv_proj<<<gqkv, 256, 0, stream>>>(keys,    WkT, bk, Kb, 0);
  k_qkv_proj<<<gqkv, 256, 0, stream>>>(values,  WvT, bv, VTb, 1);

  k_flash_attn<<<dim3(S_ / 128, B_ * H_), 256, 0, stream>>>(Qb, Kb, VTb, ATT);

  k_out_proj<<<dim3((B_ * S_) / 128, D_ / 64), 256, 0, stream>>>(ATT, WpT, bp, out);
}
